// MultiPositiveContrastiveLoss_13683765805397
// MI455X (gfx1250) — matrix operand order repaired and checked by reading, not compiled, not run on hardware
//
#include <hip/hip_runtime.h>
#include <hip/hip_bf16.h>

typedef __attribute__((ext_vector_type(16))) _Float16 v16h;
typedef __attribute__((ext_vector_type(8)))  float    v8f;

#define N_ROWS 8192
#define DIM    1024
#define NLAB   512
#define MBLK   64                  // rows per workgroup block
#define NBLK   (N_ROWS / MBLK)     // 128 row blocks
#define INV_T  14.285714285714286f // 1 / 0.07
#define EPSF   1e-8f

// LDS A-block row stride: 129 uint4 = 2064 B = 516 dwords; 516 % 64 == 4
// banks shift per row -> the 16 A-fragment rows of a wave cover all 64 banks
// (optimal 2 accesses/bank for a b128 across 32 lanes), and addresses stay
// AFFINE in the k-step so ds_load_b128 uses immediate offsets (no per-step
// VALU address math, unlike an XOR swizzle).
#define ASTRQ 129
// LDS bytes: 64 * 129 * 16 = 132096 (<= gfx1250 workgroup LDS limit)

union FragAB { v16h h; uint4 q[2]; };
union Pack4  { _Float16 h[4]; uint2 u; };

// ---------------------------------------------------------------- init ----
__global__ void init_kernel(float* __restrict__ Tot, float* __restrict__ Pos,
                            int* __restrict__ counts, float* __restrict__ acc) {
    int i = blockIdx.x * blockDim.x + threadIdx.x;
    if (i < N_ROWS) { Tot[i] = 0.0f; Pos[i] = 0.0f; }
    if (i < NLAB)   counts[i] = 0;
    if (i < 2)      acc[i] = 0.0f;
}

// ---------------------------------------------------- normalize to f16 ----
__global__ void __launch_bounds__(256)
normalize_kernel(const float* __restrict__ emb, _Float16* __restrict__ xn) {
    int row = blockIdx.x;
    int tid = threadIdx.x;
    const float4* rp = (const float4*)(emb + (size_t)row * DIM);
    float4 v = rp[tid];
    float ss = v.x * v.x + v.y * v.y + v.z * v.z + v.w * v.w;
    for (int m = 1; m < 32; m <<= 1) ss += __shfl_xor(ss, m, 32);
    __shared__ float red[8];
    __shared__ float s_scale;
    int wid = tid >> 5, lane = tid & 31;
    if (lane == 0) red[wid] = ss;
    __syncthreads();
    if (tid == 0) {
        float t = 0.0f;
        for (int i = 0; i < 8; ++i) t += red[i];
        s_scale = 1.0f / fmaxf(sqrtf(t), EPSF);
    }
    __syncthreads();
    float sc = s_scale;
    Pack4 p;
    p.h[0] = (_Float16)(v.x * sc);
    p.h[1] = (_Float16)(v.y * sc);
    p.h[2] = (_Float16)(v.z * sc);
    p.h[3] = (_Float16)(v.w * sc);
    ((uint2*)(xn + (size_t)row * DIM))[tid] = p.u;
}

// -------------------------------------------------------- label counts ----
__global__ void hist_kernel(const int* __restrict__ labels, int* __restrict__ counts) {
    int i = blockIdx.x * blockDim.x + threadIdx.x;
    if (i < N_ROWS) atomicAdd(&counts[labels[i]], 1);
}

// ------------------------------------------------------ main WMMA pass ----
// 2 WGs per 64-row block (split by column-tile parity). Within a WG:
//   wave w: row sub-strip s = w&3 (16 rows), column sub-groups {w>>2, w>>2+2}.
// All 8 waves walk the same 64-column block -> B dedup in WGP$ (4 waves per
// column sub-tile) and each A fragment feeds two WMMAs.
// Symmetric coverage at block granularity: J=(B+t) mod 128, t=0..63 always,
// t=64 iff B<64 -> each unordered block pair exactly once; t=0 is the full
// diagonal 64x64 block (row-side only).
__global__ void __launch_bounds__(256)
simloss_kernel(const _Float16* __restrict__ xn, const int* __restrict__ labels,
               float* __restrict__ Tot, float* __restrict__ Pos) {
    __shared__ uint4 Ash[MBLK * ASTRQ];   // 129 KB, padded rows

    const int tid     = threadIdx.x;
    const int blk     = blockIdx.x >> 1;
    const int par     = blockIdx.x & 1;
    const int rowbase = blk * MBLK;

    {   // cooperative load of the A block (padded rows)
        const uint4* g = (const uint4*)(xn + (size_t)rowbase * DIM);
        for (int i = tid; i < MBLK * 128; i += 256) {
            int row = i >> 7, cc = i & 127;
            Ash[row * ASTRQ + cc] = g[i];
        }
    }
    __syncthreads();

    const int wid  = tid >> 5;
    const int lane = tid & 31;
    const int h    = lane >> 4;      // wave half
    const int m    = lane & 15;      // A row-in-substrip / B,C column-in-tile
    const int s    = wid & 3;        // row sub-strip of this wave
    const int cg   = wid >> 2;       // column sub-group base (0 or 1)
    const int arow = s * 16 + m;

    // C layout: VGPR r, half h -> row (within sub-strip) r + 8h
    int rlab[8], grow[8];
#pragma unroll
    for (int r = 0; r < 8; ++r) {
        grow[r] = rowbase + s * 16 + r + h * 8;
        rlab[r] = labels[grow[r]];
    }

    float tot[8], pos[8];
#pragma unroll
    for (int r = 0; r < 8; ++r) { tot[r] = 0.0f; pos[r] = 0.0f; }

    const float DIAG_E = __expf(INV_T);   // exact exp(self-sim / T)
    const int   tmax   = (blk < NBLK / 2) ? (NBLK / 2 + 1) : (NBLK / 2);

    // per-thread A base (affine): chunk0 = abase + ks*4, chunk1 = abase + ks*4 + 1
    const uint4* aq    = Ash;
    const int    abase = arow * ASTRQ + 2 * h;

    auto epilogue = [&](const v8f& c, int gcol, int clab, bool diag) {
        float colT = 0.0f, colP = 0.0f;
#pragma unroll
        for (int r = 0; r < 8; ++r) {
            float e = __expf(c[r] * INV_T);
            bool  self = diag && (gcol == grow[r]);
            if (self) e = DIAG_E;
            float ep = (clab == rlab[r] && !self) ? e : 0.0f;
            tot[r] += e;  pos[r] += ep;
            colT   += e;  colP   += ep;
        }
        if (!diag) {   // symmetric contribution to the column block's rows
            colT += __shfl_xor(colT, 16, 32);
            colP += __shfl_xor(colP, 16, 32);
            if (h == 0) {
                atomicAdd(&Tot[gcol], colT);
                atomicAdd(&Pos[gcol], colP);
            }
        }
    };

    for (int t = par; t < tmax; t += 2) {
        const int  J       = (blk + t) & (NBLK - 1);
        const int  colbase = J * MBLK;
        const bool diag    = (t == 0);

        const int gcolA = colbase + cg * 16 + m;
        const int gcolB = colbase + (cg + 2) * 16 + m;
        const int clabA = labels[gcolA];
        const int clabB = labels[gcolB];
        const uint4* bqA = (const uint4*)(xn + (size_t)gcolA * DIM + h * 16);
        const uint4* bqB = (const uint4*)(xn + (size_t)gcolB * DIM + h * 16);

        v8f cA = {}, cB = {};
#pragma unroll 4
        for (int ks = 0; ks < DIM / 32; ++ks) {
            FragAB a, b0, b1;
            a.q[0]  = aq[abase + ks * 4];       // K: kb + 16h + 0..7
            a.q[1]  = aq[abase + ks * 4 + 1];   // K: kb + 16h + 8..15
            b0.q[0] = bqA[ks * 4];
            b0.q[1] = bqA[ks * 4 + 1];
            b1.q[0] = bqB[ks * 4];
            b1.q[1] = bqB[ks * 4 + 1];
            cA = __builtin_amdgcn_wmma_f32_16x16x32_f16(
                     false, a.h, false, b0.h, (short)0, cA, false, false);
            cB = __builtin_amdgcn_wmma_f32_16x16x32_f16(
                     false, a.h, false, b1.h, (short)0, cB, false, false);
        }

        epilogue(cA, gcolA, clabA, diag);
        epilogue(cB, gcolB, clabB, diag);
        __syncthreads();   // keep waves in lock-step for WGP$ B reuse
    }

    // row-side: reduce over the 16 lanes of each half, then global atomics
#pragma unroll
    for (int r = 0; r < 8; ++r) {
        float tr = tot[r], pr = pos[r];
        for (int msk = 1; msk < 16; msk <<= 1) {
            tr += __shfl_xor(tr, msk, 32);
            pr += __shfl_xor(pr, msk, 32);
        }
        if (m == 0) {
            atomicAdd(&Tot[grow[r]], tr);
            atomicAdd(&Pos[grow[r]], pr);
        }
    }
}

// -------------------------------------------------------- per-row loss ----
__global__ void __launch_bounds__(256)
rowloss_kernel(const float* __restrict__ Tot, const float* __restrict__ Pos,
               const int* __restrict__ labels, const int* __restrict__ counts,
               float* __restrict__ acc) {
    int i = blockIdx.x * blockDim.x + threadIdx.x;
    float l = 0.0f, v = 0.0f;
    if (i < N_ROWS) {
        int pcnt = counts[labels[i]] - 1;
        if (pcnt > 0) {
            l = -logf(Pos[i] / (Tot[i] + EPSF) + EPSF);
            v = 1.0f;
        }
    }
    for (int msk = 1; msk < 32; msk <<= 1) {
        l += __shfl_xor(l, msk, 32);
        v += __shfl_xor(v, msk, 32);
    }
    if ((threadIdx.x & 31) == 0 && (l != 0.0f || v != 0.0f)) {
        atomicAdd(&acc[0], l);
        atomicAdd(&acc[1], v);
    }
}

__global__ void finalize_kernel(const float* __restrict__ acc, float* __restrict__ out) {
    if (threadIdx.x == 0) {
        float nv = acc[1];
        out[0] = (nv > 0.0f) ? acc[0] / fmaxf(nv, 1.0f) : 0.0f;
    }
}

// ------------------------------------------------------------ launcher ----
extern "C" void kernel_launch(void* const* d_in, const int* in_sizes, int n_in,
                              void* d_out, int out_size, void* d_ws, size_t ws_size,
                              hipStream_t stream) {
    const float* emb    = (const float*)d_in[0];
    const int*   labels = (const int*)d_in[1];
    float*       out    = (float*)d_out;

    char* ws = (char*)d_ws;
    const size_t XN_BYTES = (size_t)N_ROWS * DIM * sizeof(_Float16); // 16 MB
    _Float16* xn     = (_Float16*)ws;
    float*    Tot    = (float*)(ws + XN_BYTES);
    float*    Pos    = (float*)(ws + XN_BYTES + (size_t)N_ROWS * 4);
    int*      counts = (int*)  (ws + XN_BYTES + (size_t)N_ROWS * 8);
    float*    acc    = (float*)(ws + XN_BYTES + (size_t)N_ROWS * 8 + 4096);

    init_kernel<<<N_ROWS / 256, 256, 0, stream>>>(Tot, Pos, counts, acc);
    normalize_kernel<<<N_ROWS, 256, 0, stream>>>(emb, xn);
    hist_kernel<<<N_ROWS / 256, 256, 0, stream>>>(labels, counts);
    simloss_kernel<<<2 * NBLK, 256, 0, stream>>>(xn, labels, Tot, Pos);
    rowloss_kernel<<<N_ROWS / 256, 256, 0, stream>>>(Tot, Pos, labels, counts, acc);
    finalize_kernel<<<1, 32, 0, stream>>>(acc, out);
}
